// SelectiveSSM_44255343018513
// MI455X (gfx1250) — compile-verified, matrix operand order checked by reading, not run on hardware
//
#include <hip/hip_runtime.h>

typedef __attribute__((ext_vector_type(16))) __bf16 v16bf;
typedef __attribute__((ext_vector_type(8)))  float  v8f;
typedef __attribute__((ext_vector_type(4)))  int    v4i;

// Problem sizes (compile-time)
constexpr int Bn   = 4;
constexpr int Ln   = 1024;
constexpr int Dn   = 2048;   // D_INNER
constexpr int Ns   = 16;     // D_STATE
constexpr int Rr   = 128;    // DT_RANK
constexpr int K1   = Rr + 2 * Ns;   // 160 output cols of GEMM1
constexpr int Mtot = Bn * Ln;       // 4096

// ---------------------------------------------------------------------------
// Async global->LDS (CDNA5 GLOBAL_LOAD_ASYNC_TO_LDS_B128, ASYNCcnt-tracked).
// Builtin signature (from hipcc diagnostic): (v4i AS1*, v4i AS3*, imm, imm).
// Falls back to a synchronous float4 copy if the builtins are absent.
// ---------------------------------------------------------------------------
#if __has_builtin(__builtin_amdgcn_global_load_async_to_lds_b128) && \
    __has_builtin(__builtin_amdgcn_s_wait_asynccnt)
#define HAVE_ASYNC_LDS 1
#endif

static __device__ __forceinline__ void g2l_b128(void* lds, const void* g) {
#ifdef HAVE_ASYNC_LDS
    __builtin_amdgcn_global_load_async_to_lds_b128(
        (__attribute__((address_space(1))) v4i*)(const v4i*)g,
        (__attribute__((address_space(3))) v4i*)(v4i*)lds, 0, 0);
#else
    *(float4*)lds = *(const float4*)g;
#endif
}
static __device__ __forceinline__ void wait_async_lds() {
#ifdef HAVE_ASYNC_LDS
    __builtin_amdgcn_s_wait_asynccnt(0);
#endif
}

static __device__ __forceinline__ unsigned short f2bf(float f) {
    union { float f; unsigned u; } v; v.f = f;
    unsigned r = v.u + 0x7FFFu + ((v.u >> 16) & 1u);   // round-to-nearest-even
    return (unsigned short)(r >> 16);
}

// K-pair column for VGPR j of a 16x32 bf16 WMMA fragment (ISA 7.12.2 layout):
// lanes 0-15: VGPR0-3 -> K=0..7 pairs, VGPR4-7 -> K=16..23; lanes 16-31: +8.
static __device__ __forceinline__ int kpair(int j, int half) {
    return ((j < 4) ? (2 * j) : (16 + 2 * (j - 4))) + 8 * half;
}

// ---------------------------------------------------------------------------
// Kernel 0: one-time fp32 -> bf16 conversion of W_x and W_dt (L2-resident).
// ---------------------------------------------------------------------------
__global__ void convert_weights(const float* __restrict__ Wx,
                                const float* __restrict__ Wdt,
                                unsigned short* __restrict__ Wx_bf,
                                unsigned short* __restrict__ Wdt_bf) {
    int i = blockIdx.x * blockDim.x + threadIdx.x;
    const int nWx  = K1 * Dn;     // 160*2048
    const int nWdt = Dn * Rr;     // 2048*128
    if (i < nWx)                 Wx_bf[i]        = f2bf(Wx[i]);
    else if (i < nWx + nWdt)     Wdt_bf[i - nWx] = f2bf(Wdt[i - nWx]);
}

// ---------------------------------------------------------------------------
// Kernel 1: xz = x @ W_x^T   (M=4096, K=2048, N=160), bf16 WMMA, fp32 acc.
// Each block: 64 M-rows x all 160 N. 4 waves; wave w owns M-tile w, all 10
// N-tiles. x converted to bf16 into LDS on the fly (single HBM read of x);
// W tile staged via async-to-LDS. 64 blocks for memory-level parallelism.
// Output split: n<128 -> bf16 dt_raw, n>=128 -> f32 b/c slices.
// ---------------------------------------------------------------------------
__global__ __launch_bounds__(128) void gemm_xz(
    const float* __restrict__ x, const unsigned short* __restrict__ Wx_bf,
    unsigned short* __restrict__ dtr_bf, float* __restrict__ bc) {
    constexpr int LDA = 40, LDB = 40;           // short strides, 16B-aligned rows
    __shared__ unsigned short sA[64 * LDA];     // 64  x 32 bf16 x-tile
    __shared__ unsigned short sB[K1 * LDB];     // 160 x 32 bf16 W-tile

    const int tid  = threadIdx.x;
    const int wave = tid >> 5, lane = tid & 31;
    const int half = lane >> 4, l16 = lane & 15;
    const int m0   = blockIdx.x * 64;

    v8f acc[10];
    #pragma unroll
    for (int t = 0; t < 10; ++t) acc[t] = (v8f)0.0f;

    for (int k0 = 0; k0 < Dn; k0 += 32) {
        // stage W_x bf16 block 160x32 via async b128 (4 segs of 8 shorts/row)
        #pragma unroll
        for (int j = 0; j < 5; ++j) {
            int idx = tid + 128 * j;            // 0..639
            int r = idx >> 2, seg = idx & 3;
            g2l_b128(&sB[r * LDB + seg * 8],
                     Wx_bf + (size_t)r * Dn + k0 + seg * 8);
        }
        // stage x block 64x32 f32 -> bf16 (each thread: one half-row, 4xfloat4)
        {
            int r  = tid >> 1;                  // 0..63
            int q0 = (tid & 1) * 4;
            const float4* src = (const float4*)(x + (size_t)(m0 + r) * Dn + k0);
            #pragma unroll
            for (int j = 0; j < 4; ++j) {
                float4 v = src[q0 + j];
                unsigned short* dst = &sA[r * LDA + (q0 + j) * 4];
                dst[0] = f2bf(v.x); dst[1] = f2bf(v.y);
                dst[2] = f2bf(v.z); dst[3] = f2bf(v.w);
            }
        }
        wait_async_lds();
        __syncthreads();

        union { v16bf v; unsigned u[8]; } afr;
        const int arow = wave * 16 + l16;
        #pragma unroll
        for (int j = 0; j < 8; ++j)
            afr.u[j] = *(const unsigned*)&sA[arow * LDA + kpair(j, half)];

        #pragma unroll
        for (int nt = 0; nt < 10; ++nt) {
            union { v16bf v; unsigned u[8]; } bfr;
            const int nrow = nt * 16 + l16;
            #pragma unroll
            for (int j = 0; j < 8; ++j)
                bfr.u[j] = *(const unsigned*)&sB[nrow * LDB + kpair(j, half)];
            acc[nt] = __builtin_amdgcn_wmma_f32_16x16x32_bf16(
                false, afr.v, false, bfr.v, (short)0, acc[nt], false, false);
        }
        __syncthreads();
    }

    // epilogue: C layout VGPR r -> M = r + 8*half, N = lane&15
    #pragma unroll
    for (int nt = 0; nt < 10; ++nt) {
        int n = nt * 16 + l16;
        #pragma unroll
        for (int r = 0; r < 8; ++r) {
            int m = m0 + wave * 16 + r + 8 * half;
            float v = acc[nt][r];
            if (n < Rr) dtr_bf[(size_t)m * Rr + n] = f2bf(v);
            else        bc[(size_t)m * (2 * Ns) + (n - Rr)] = v;
        }
    }
}

// ---------------------------------------------------------------------------
// Kernel 2: dt = softplus(dt_raw @ W_dt^T + b_dt)  (M=4096, K=128, N=2048).
// Block tile 32M x 128N, whole K resident in LDS (async-staged).
// 8 waves x 2 tiles each; 2048 blocks.
// ---------------------------------------------------------------------------
__global__ __launch_bounds__(256) void gemm_dt(
    const unsigned short* __restrict__ dtr_bf,
    const unsigned short* __restrict__ Wdt_bf,
    const float* __restrict__ b_dt, float* __restrict__ dt_out) {
    constexpr int LDK = 136;                    // short stride, 16B-aligned rows
    __shared__ unsigned short sA[32 * LDK];     // 32  x 128 bf16
    __shared__ unsigned short sB[128 * LDK];    // 128 x 128 bf16

    const int tid  = threadIdx.x;
    const int wave = tid >> 5, lane = tid & 31;
    const int half = lane >> 4, l16 = lane & 15;
    const int m0   = blockIdx.x * 32;
    const int n0   = blockIdx.y * 128;

    #pragma unroll
    for (int j = 0; j < 2; ++j) {               // A: 32 rows x 16 b128 segs
        int idx = tid + 256 * j;
        int r = idx >> 4, seg = idx & 15;
        g2l_b128(&sA[r * LDK + seg * 8],
                 dtr_bf + (size_t)(m0 + r) * Rr + seg * 8);
    }
    #pragma unroll
    for (int j = 0; j < 8; ++j) {               // B: 128 rows x 16 b128 segs
        int idx = tid + 256 * j;
        int r = idx >> 4, seg = idx & 15;
        g2l_b128(&sB[r * LDK + seg * 8],
                 Wdt_bf + (size_t)(n0 + r) * Rr + seg * 8);
    }
    wait_async_lds();
    __syncthreads();

    const int mt = wave >> 2;                   // 0..1
    const int ng = wave & 3;                    // 0..3 (2 N-tiles each)
    v8f acc[2]; acc[0] = (v8f)0.0f; acc[1] = (v8f)0.0f;

    #pragma unroll
    for (int kc = 0; kc < 4; ++kc) {
        union { v16bf v; unsigned u[8]; } afr;
        const int arow = mt * 16 + l16;
        #pragma unroll
        for (int j = 0; j < 8; ++j)
            afr.u[j] = *(const unsigned*)&sA[arow * LDK + kc * 32 + kpair(j, half)];
        #pragma unroll
        for (int t = 0; t < 2; ++t) {
            union { v16bf v; unsigned u[8]; } bfr;
            const int nrow = ng * 32 + t * 16 + l16;
            #pragma unroll
            for (int j = 0; j < 8; ++j)
                bfr.u[j] = *(const unsigned*)&sB[nrow * LDK + kc * 32 + kpair(j, half)];
            acc[t] = __builtin_amdgcn_wmma_f32_16x16x32_bf16(
                false, afr.v, false, bfr.v, (short)0, acc[t], false, false);
        }
    }

    #pragma unroll
    for (int t = 0; t < 2; ++t) {
        int n = n0 + ng * 32 + t * 16 + l16;
        float bb = b_dt[n];
        #pragma unroll
        for (int r = 0; r < 8; ++r) {
            int m = m0 + mt * 16 + r + 8 * half;
            float z  = acc[t][r] + bb;
            float sp = (z > 15.0f) ? z : __logf(1.0f + __expf(z));  // softplus
            dt_out[(size_t)m * Dn + n] = sp;
        }
    }
}

// ---------------------------------------------------------------------------
// Kernel 3: fused selective scan. 16 lanes per (b,d) channel, one state n per
// lane -> 131K threads / 4096 waves. da/dbx recomputed in-register; y via
// shfl_xor butterfly. dt/x/b/c async-staged through LDS in 64-step chunks.
// ---------------------------------------------------------------------------
__global__ __launch_bounds__(256) void scan_fused(
    const float* __restrict__ x, const float* __restrict__ dt,
    const float* __restrict__ bc, const float* __restrict__ A_log,
    const float* __restrict__ D_param, float* __restrict__ out) {
    constexpr int TC = 64;
    __shared__ float sdt[TC][16];
    __shared__ float sx [TC][16];
    __shared__ float sb [TC][16];
    __shared__ float sc [TC][16];

    const int tid = threadIdx.x;
    const int blocksPerB = Dn / 16;                     // 128
    const int b  = blockIdx.x / blocksPerB;
    const int d0 = (blockIdx.x % blocksPerB) * 16;
    const int ch = tid >> 4;                            // channel in block
    const int n  = tid & 15;                            // state index
    const int d  = d0 + ch;

    const float a  = -__expf(A_log[(size_t)d * Ns + n]);
    const float Dp = D_param[d];
    float h = 0.0f;

    for (int l0 = 0; l0 < Ln; l0 += TC) {
        __syncthreads();                                // protect prior chunk
        {   // dt & x chunks: 64x16 floats = 256 b128 each; 1 per thread
            int t = tid >> 2, c4 = (tid & 3) * 4;
            size_t g = (size_t)(b * Ln + l0 + t) * Dn + d0 + c4;
            g2l_b128(&sdt[t][c4], dt + g);
            g2l_b128(&sx [t][c4], x  + g);
        }
        #pragma unroll
        for (int j = 0; j < 2; ++j) {                   // b,c: 64x32 = 512 b128
            int idx = tid + 256 * j;
            int t = idx >> 3, c4 = idx & 7;
            const float* g = bc + (size_t)(b * Ln + l0 + t) * (2 * Ns) + c4 * 4;
            if (c4 < 4) g2l_b128(&sb[t][c4 * 4],       g);
            else        g2l_b128(&sc[t][(c4 - 4) * 4], g);
        }
        wait_async_lds();
        __syncthreads();

        for (int t = 0; t < TC; ++t) {
            float dtv = sdt[t][ch];                     // LDS broadcast
            float xv  = sx [t][ch];
            float da  = __expf(dtv * a);
            h = da * h + (dtv * sb[t][n]) * xv;         // h = da*h + dt*b*x
            float p = h * sc[t][n];                     // y = sum_n h*c
            p += __shfl_xor(p, 1, 16);
            p += __shfl_xor(p, 2, 16);
            p += __shfl_xor(p, 4, 16);
            p += __shfl_xor(p, 8, 16);
            if (n == 0)
                out[(size_t)(b * Ln + l0 + t) * Dn + d] = p + xv * Dp;
        }
    }
}

// ---------------------------------------------------------------------------
extern "C" void kernel_launch(void* const* d_in, const int* in_sizes, int n_in,
                              void* d_out, int out_size, void* d_ws, size_t ws_size,
                              hipStream_t stream) {
    const float* x     = (const float*)d_in[0];
    const float* W_x   = (const float*)d_in[1];
    const float* W_dt  = (const float*)d_in[2];
    const float* b_dt  = (const float*)d_in[3];
    const float* A_log = (const float*)d_in[4];
    const float* D_par = (const float*)d_in[5];
    float* out = (float*)d_out;

    char* w = (char*)d_ws;
    auto carve = [&](size_t bytes) -> char* {
        char* p = w; w += (bytes + 255) & ~(size_t)255; return p;
    };
    unsigned short* Wx_bf  = (unsigned short*)carve((size_t)K1 * Dn * 2);   // 640 KB
    unsigned short* Wdt_bf = (unsigned short*)carve((size_t)Dn * Rr * 2);   // 512 KB
    unsigned short* dtr_bf = (unsigned short*)carve((size_t)Mtot * Rr * 2); // 1 MB
    float*          bc     = (float*)carve((size_t)Mtot * 2 * Ns * 4);      // 512 KB
    float*          dtws   = (float*)carve((size_t)Mtot * Dn * 4);          // 32 MB

    const int nconv = K1 * Dn + Dn * Rr;
    hipLaunchKernelGGL(convert_weights, dim3((nconv + 255) / 256), dim3(256), 0, stream,
                       W_x, W_dt, Wx_bf, Wdt_bf);
    hipLaunchKernelGGL(gemm_xz, dim3(Mtot / 64), dim3(128), 0, stream,
                       x, Wx_bf, dtr_bf, bc);
    hipLaunchKernelGGL(gemm_dt, dim3(Mtot / 32, Dn / 128), dim3(256), 0, stream,
                       dtr_bf, Wdt_bf, b_dt, dtws);
    hipLaunchKernelGGL(scan_fused, dim3(Bn * Dn / 16), dim3(256), 0, stream,
                       x, dtws, bc, A_log, D_par, out);
}
